// Encoder_73598559584739
// MI455X (gfx1250) — matrix operand order repaired and checked by reading, not compiled, not run on hardware
//
#include <hip/hip_runtime.h>

// ---------------------------------------------------------------------------
// CDNA5 (gfx1250) GraphSAGE encoder
//   - bf16 WMMA GEMMs (v_wmma_f32_16x16x32_bf16), dual-operand fusion for
//     mean@Wl + root@Wr + b
//   - bf16 operand streams in global memory (halves HBM traffic); fp32 only
//     for the atomic scatter-mean accumulator
//   - async global->LDS tile staging (ASYNCcnt) with double buffering
// ---------------------------------------------------------------------------

typedef __attribute__((ext_vector_type(16))) __bf16 v16bf;
typedef __attribute__((ext_vector_type(8)))  float  v8f;
typedef unsigned short u16;
typedef unsigned int   u32;

#define BM 64
#define BN 64
#define BK 32
#define LDT 40   // LDS row stride (ushorts): 80B -> 16B aligned, bank-spread

#if defined(__HIP_DEVICE_COMPILE__) && __has_builtin(__builtin_amdgcn_global_load_async_to_lds_b128)
#define HAVE_ASYNC 1
typedef __attribute__((__vector_size__(16))) int v4i;
typedef __attribute__((address_space(1))) v4i gv4i;   // global int4
typedef __attribute__((address_space(3))) v4i lv4i;   // LDS int4
#else
#define HAVE_ASYNC 0
#endif

__device__ __forceinline__ u16 f2bf(float f) {  // fp32 -> bf16 RNE
    u32 u = __float_as_uint(f);
    u += 0x7FFFu + ((u >> 16) & 1u);
    return (u16)(u >> 16);
}
__device__ __forceinline__ u32 pk2(float a, float b) {
    return (u32)f2bf(a) | ((u32)f2bf(b) << 16);
}
__device__ __forceinline__ float bflo(u32 u) { return __uint_as_float(u << 16); }
__device__ __forceinline__ float bfhi(u32 u) { return __uint_as_float(u & 0xFFFF0000u); }

// copy 16B global -> LDS (async when the toolchain exposes the builtin)
__device__ __forceinline__ void cp16(u16* l, const u16* g) {
#if HAVE_ASYNC
    __builtin_amdgcn_global_load_async_to_lds_b128((gv4i*)g, (lv4i*)l, 0, 0);
#else
    *(uint4*)l = *(const uint4*)g;
#endif
}
__device__ __forceinline__ void async_fence() {
#if HAVE_ASYNC
#if __has_builtin(__builtin_amdgcn_s_wait_asynccnt)
    __builtin_amdgcn_s_wait_asynccnt(0);
#else
    asm volatile("s_wait_asynccnt 0x0" ::: "memory");
#endif
#endif
}

union FragBF {
    uint4 q[2];
    v16bf v;
};

// C = act( Af*rowscale @ Wft^T  +  Ab @ Wbt^T  + bias )
//   Af : fp32 [M,K] (optional; the scatter-mean accumulator), rowscale fp32[M]
//   Ab : bf16 [M,K] (always present)
//   Wft/Wbt : bf16 pre-transposed weights [N,K]
//   Output: Cf fp32 [M,N] or Cb bf16 [M,N] (exactly one non-null)
// Assumes M%64==0, N%64==0, K%32==0 (holds: M=80000, N/K in {256,128}).
__global__ __launch_bounds__(256)
void k_gemm(const float* __restrict__ Af, const float* __restrict__ rowscale,
            const u16* __restrict__ Ab,
            const u16* __restrict__ Wft, const u16* __restrict__ Wbt,
            const float* __restrict__ bias,
            float* __restrict__ Cf, u16* __restrict__ Cb,
            int M, int K, int N, int do_relu) {
    __shared__ u16 sAf[2][BM * LDT];
    __shared__ u16 sAb[2][BM * LDT];
    __shared__ u16 sWf[2][BN * LDT];
    __shared__ u16 sWb[2][BN * LDT];   // 8 * 5120B = 40KB

    const int t    = threadIdx.x;
    const int lane = t & 31;
    const int wave = t >> 5;
    const int wm   = wave & 3;          // M sub-tile: wm*16
    const int wn   = wave >> 2;         // N sub-tile: wn*32
    const int mblk = blockIdx.x * BM;
    const int nblk = blockIdx.y * BN;
    const bool dual = (Af != nullptr);

    // tile staging: 256 threads x 16B; cr = row (0..63), cc = K-chunk (0,8,16,24)
    const int cr = t >> 2;
    const int cc = (t & 3) * 8;
    const u16*   gAb = Ab  + (size_t)(mblk + cr) * K + cc;
    const u16*   gWb = Wbt + (size_t)(nblk + cr) * K + cc;
    const u16*   gWf = dual ? (Wft + (size_t)(nblk + cr) * K + cc) : nullptr;
    const float* gAf = dual ? (Af  + (size_t)(mblk + cr) * K + cc) : nullptr;
    const int    loff = cr * LDT + cc;
    const float  rs = (dual && rowscale) ? rowscale[mblk + cr] : 1.0f;

    auto stage = [&](int buf, int k0) {
        cp16(&sAb[buf][loff], gAb + k0);
        cp16(&sWb[buf][loff], gWb + k0);
        if (dual) {
            cp16(&sWf[buf][loff], gWf + k0);
            const float* g = gAf + k0;               // fp32 + scale -> bf16
            float4 p = *(const float4*)(g);
            float4 q = *(const float4*)(g + 4);
            u32* d = (u32*)&sAf[buf][loff];
            d[0] = pk2(p.x * rs, p.y * rs);
            d[1] = pk2(p.z * rs, p.w * rs);
            d[2] = pk2(q.x * rs, q.y * rs);
            d[3] = pk2(q.z * rs, q.w * rs);
        }
    };

    // fragment indices
    // A layout: lanes 0-15 K=0..15, lanes 16-31 K=16..31
    const int arow = (wm * 16 + (lane & 15)) * LDT;
    const int kg   = (lane >> 4) * 16;
    // B ([n][k] in LDS): lane = n; lanes 0-15 K=0..15, lanes 16-31 K=16..31
    const int nrow0 = (wn * 32 + (lane & 15)) * LDT;
    const int nrow1 = nrow0 + 16 * LDT;
    const int kb    = (lane >> 4) * 16;

    v8f acc0 = {}, acc1 = {};
    const int nsteps = K / BK;

    stage(0, 0);
    for (int s = 0; s < nsteps; ++s) {
        async_fence();
        __syncthreads();
        if (s + 1 < nsteps) stage((s + 1) & 1, (s + 1) * BK);
        const u16* bAb = sAb[s & 1];
        const u16* bWb = sWb[s & 1];

        FragBF a, b0, b1;
        a.q[0]  = *(const uint4*)(bAb + arow + kg);
        a.q[1]  = *(const uint4*)(bAb + arow + kg + 8);
        b0.q[0] = *(const uint4*)(bWb + nrow0 + kb);
        b0.q[1] = *(const uint4*)(bWb + nrow0 + kb + 8);
        b1.q[0] = *(const uint4*)(bWb + nrow1 + kb);
        b1.q[1] = *(const uint4*)(bWb + nrow1 + kb + 8);
        acc0 = __builtin_amdgcn_wmma_f32_16x16x32_bf16(false, a.v, false, b0.v,
                                                       (short)0, acc0, false, false);
        acc1 = __builtin_amdgcn_wmma_f32_16x16x32_bf16(false, a.v, false, b1.v,
                                                       (short)0, acc1, false, false);
        if (dual) {
            const u16* bAf = sAf[s & 1];
            const u16* bWf = sWf[s & 1];
            FragBF af, c0, c1;
            af.q[0] = *(const uint4*)(bAf + arow + kg);
            af.q[1] = *(const uint4*)(bAf + arow + kg + 8);
            c0.q[0] = *(const uint4*)(bWf + nrow0 + kb);
            c0.q[1] = *(const uint4*)(bWf + nrow0 + kb + 8);
            c1.q[0] = *(const uint4*)(bWf + nrow1 + kb);
            c1.q[1] = *(const uint4*)(bWf + nrow1 + kb + 8);
            acc0 = __builtin_amdgcn_wmma_f32_16x16x32_bf16(false, af.v, false, c0.v,
                                                           (short)0, acc0, false, false);
            acc1 = __builtin_amdgcn_wmma_f32_16x16x32_bf16(false, af.v, false, c1.v,
                                                           (short)0, acc1, false, false);
        }
    }

    // epilogue; D layout: VGPR r -> M = r + 8*(lane>=16); lane&15 -> N
    const int   ng0 = nblk + wn * 32 + (lane & 15);
    const int   ng1 = ng0 + 16;
    const float bb0 = bias ? bias[ng0] : 0.0f;
    const float bb1 = bias ? bias[ng1] : 0.0f;
    const int   mb  = mblk + wm * 16 + (lane >> 4) * 8;
#pragma unroll
    for (int r = 0; r < 8; ++r) {
        float v0 = acc0[r] + bb0;
        float v1 = acc1[r] + bb1;
        if (do_relu) { v0 = fmaxf(v0, 0.0f); v1 = fmaxf(v1, 0.0f); }
        if (Cf) {
            Cf[(size_t)(mb + r) * N + ng0] = v0;
            Cf[(size_t)(mb + r) * N + ng1] = v1;
        } else {
            Cb[(size_t)(mb + r) * N + ng0] = f2bf(v0);
            Cb[(size_t)(mb + r) * N + ng1] = f2bf(v1);
        }
    }
}

// fp32 -> bf16 bulk convert (8 elems/thread)
__global__ __launch_bounds__(256)
void k_cvt(const float* __restrict__ in, u16* __restrict__ out, int n8) {
    const int i = blockIdx.x * blockDim.x + threadIdx.x;
    if (i >= n8) return;
    const float* g = in + (size_t)i * 8;
    float4 p = *(const float4*)(g);
    float4 q = *(const float4*)(g + 4);
    uint4 o;
    o.x = pk2(p.x, p.y); o.y = pk2(p.z, p.w);
    o.z = pk2(q.x, q.y); o.w = pk2(q.z, q.w);
    *(uint4*)(out + (size_t)i * 8) = o;
}

// weight pre-transpose + convert: Wt[n*K+k] = bf16(W[k*N+n])
__global__ __launch_bounds__(256)
void k_wt(const float* __restrict__ W, u16* __restrict__ Wt, int K, int N) {
    const int i = blockIdx.x * blockDim.x + threadIdx.x;
    if (i >= N * K) return;
    const int n = i / K;
    const int k = i - n * K;
    Wt[i] = f2bf(W[(size_t)k * N + n]);
}

// scatter-mean accumulate: agg[dst] += bf16(xp[src]); cnt[dst] += 1
// one thread per (edge, 8-bf16 chunk); agg (80MB) stays L2-resident on MI455X.
__global__ __launch_bounds__(256)
void k_scatter(const int* __restrict__ src, const int* __restrict__ dst,
               const u16* __restrict__ xp, float* __restrict__ agg,
               float* __restrict__ cnt, int E, int F) {
    const int cpe = F >> 3;
    const int tid = blockIdx.x * blockDim.x + threadIdx.x;
    if (tid >= E * cpe) return;
    const int e = tid / cpe;
    const int c = tid - e * cpe;
    const int s = src[e];
    const int d = dst[e];
    const uint4 v = *(const uint4*)(xp + (size_t)s * F + c * 8);
    float* p = agg + (size_t)d * F + c * 8;
    atomicAdd(p + 0, bflo(v.x)); atomicAdd(p + 1, bfhi(v.x));
    atomicAdd(p + 2, bflo(v.y)); atomicAdd(p + 3, bfhi(v.y));
    atomicAdd(p + 4, bflo(v.z)); atomicAdd(p + 5, bfhi(v.z));
    atomicAdd(p + 6, bflo(v.w)); atomicAdd(p + 7, bfhi(v.w));
    if (c == 0) atomicAdd(cnt + d, 1.0f);
}

__global__ __launch_bounds__(256)
void k_invcnt(const float* __restrict__ cnt, float* __restrict__ icnt, int n) {
    const int i = blockIdx.x * blockDim.x + threadIdx.x;
    if (i < n) icnt[i] = 1.0f / fmaxf(cnt[i], 1.0f);
}

// local_feat[b,j,:] = h[b*num_nodes + map_id[b,j], :]  (bs is a device scalar)
__global__ __launch_bounds__(256)
void k_gather(const float* __restrict__ h, const int* __restrict__ map_id,
              const int* __restrict__ bs_ptr, float* __restrict__ out,
              int total_map, int Nnodes, int REP) {
    const int tid = blockIdx.x * blockDim.x + threadIdx.x;
    if (tid >= total_map * REP) return;
    const int row = tid / REP;
    const int f   = tid - row * REP;
    const int bs  = *bs_ptr;
    const int Kt  = total_map / bs;
    const int nn  = Nnodes / bs;
    const int b   = row / Kt;
    out[(size_t)row * REP + f] = h[((size_t)b * nn + map_id[row]) * REP + f];
}

extern "C" void kernel_launch(void* const* d_in, const int* in_sizes, int n_in,
                              void* d_out, int out_size, void* d_ws, size_t ws_size,
                              hipStream_t stream) {
    const float* x    = (const float*)d_in[0];
    const int*   ei   = (const int*)d_in[1];
    const int*   mapi = (const int*)d_in[2];
    const int*   bs_p = (const int*)d_in[3];
    const float* Wp1  = (const float*)d_in[4];
    const float* bp1  = (const float*)d_in[5];
    const float* Wl1  = (const float*)d_in[6];
    const float* bl1  = (const float*)d_in[7];
    const float* Wr1  = (const float*)d_in[8];
    const float* Wp2  = (const float*)d_in[9];
    const float* bp2  = (const float*)d_in[10];
    const float* Wl2  = (const float*)d_in[11];
    const float* bl2  = (const float*)d_in[12];
    const float* Wr2  = (const float*)d_in[13];
    const float* Wlin = (const float*)d_in[14];
    const float* blin = (const float*)d_in[15];

    const int F   = in_sizes[5];        // 256
    const int HID = in_sizes[7];        // 256
    const int REP = in_sizes[12];       // 128
    const int Nn  = in_sizes[0] / F;    // 80000 (divisible by 64)
    const int E   = in_sizes[1] / 2;    // 1,280,000
    const int TM  = in_sizes[2];        // 40000
    const int* src = ei;
    const int* dst = ei + E;

    // ---- workspace layout: fp32 region first, then bf16 region ----
    float* fws  = (float*)d_ws;
    size_t fo   = 0;
    float* agg  = fws + fo; fo += (size_t)Nn * F;
    float* cnt  = fws + fo; fo += (size_t)Nn;
    float* icnt = fws + fo; fo += (size_t)Nn;
    u16*   bws  = (u16*)(fws + fo);
    size_t bo   = 0;
    u16* xb   = bws + bo; bo += (size_t)Nn * F;     // bf16(x)
    u16* xpb  = bws + bo; bo += (size_t)Nn * F;     // projected messages
    u16* h1b  = bws + bo; bo += (size_t)Nn * HID;
    u16* h2b  = bws + bo; bo += (size_t)Nn * REP;
    u16* Wp1t = bws + bo; bo += (size_t)F * F;
    u16* Wl1t = bws + bo; bo += (size_t)F * HID;
    u16* Wr1t = bws + bo; bo += (size_t)F * HID;
    u16* Wp2t = bws + bo; bo += (size_t)HID * HID;
    u16* Wl2t = bws + bo; bo += (size_t)HID * REP;
    u16* Wr2t = bws + bo; bo += (size_t)HID * REP;
    u16* Wlnt = bws + bo; bo += (size_t)REP * REP;

    float* lfeat = (float*)d_out;                    // [TM, REP]
    float* hout  = (float*)d_out + (size_t)TM * REP; // [Nn, REP]

    const dim3 blk(256);
    auto cdiv = [](size_t a, size_t b) { return (unsigned)((a + b - 1) / b); };

    // ---- one-time stream conversions ----
    k_cvt<<<cdiv((size_t)Nn * F / 8, 256), blk, 0, stream>>>(x, xb, Nn * F / 8);
    k_wt<<<cdiv((size_t)F * F, 256),     blk, 0, stream>>>(Wp1,  Wp1t, F,   F);
    k_wt<<<cdiv((size_t)F * HID, 256),   blk, 0, stream>>>(Wl1,  Wl1t, F,   HID);
    k_wt<<<cdiv((size_t)F * HID, 256),   blk, 0, stream>>>(Wr1,  Wr1t, F,   HID);
    k_wt<<<cdiv((size_t)HID * HID, 256), blk, 0, stream>>>(Wp2,  Wp2t, HID, HID);
    k_wt<<<cdiv((size_t)HID * REP, 256), blk, 0, stream>>>(Wl2,  Wl2t, HID, REP);
    k_wt<<<cdiv((size_t)HID * REP, 256), blk, 0, stream>>>(Wr2,  Wr2t, HID, REP);
    k_wt<<<cdiv((size_t)REP * REP, 256), blk, 0, stream>>>(Wlin, Wlnt, REP, REP);

    // ---- layer 1 ----
    k_gemm<<<dim3(Nn / 64, F / 64), blk, 0, stream>>>(
        nullptr, nullptr, xb, nullptr, Wp1t, bp1, nullptr, xpb, Nn, F, F, 1);
    (void)hipMemsetAsync(agg, 0, (size_t)Nn * F * sizeof(float), stream);
    (void)hipMemsetAsync(cnt, 0, (size_t)Nn * sizeof(float), stream);
    k_scatter<<<cdiv((size_t)E * (F / 8), 256), blk, 0, stream>>>(src, dst, xpb, agg, cnt, E, F);
    k_invcnt<<<cdiv(Nn, 256), blk, 0, stream>>>(cnt, icnt, Nn);
    k_gemm<<<dim3(Nn / 64, HID / 64), blk, 0, stream>>>(
        agg, icnt, xb, Wl1t, Wr1t, bl1, nullptr, h1b, Nn, F, HID, 1);

    // ---- layer 2 ----
    k_gemm<<<dim3(Nn / 64, HID / 64), blk, 0, stream>>>(
        nullptr, nullptr, h1b, nullptr, Wp2t, bp2, nullptr, xpb, Nn, HID, HID, 1);
    (void)hipMemsetAsync(agg, 0, (size_t)Nn * HID * sizeof(float), stream);
    (void)hipMemsetAsync(cnt, 0, (size_t)Nn * sizeof(float), stream);
    k_scatter<<<cdiv((size_t)E * (HID / 8), 256), blk, 0, stream>>>(src, dst, xpb, agg, cnt, E, HID);
    k_invcnt<<<cdiv(Nn, 256), blk, 0, stream>>>(cnt, icnt, Nn);
    k_gemm<<<dim3(Nn / 64, REP / 64), blk, 0, stream>>>(
        agg, icnt, h1b, Wl2t, Wr2t, bl2, nullptr, h2b, Nn, HID, REP, 1);

    // ---- final linear straight into d_out's h region ----
    k_gemm<<<dim3(Nn / 64, REP / 64), blk, 0, stream>>>(
        nullptr, nullptr, h2b, nullptr, Wlnt, blin, hout, nullptr, Nn, REP, REP, 0);

    // ---- kriging-target gather ----
    k_gather<<<cdiv((size_t)TM * REP, 256), blk, 0, stream>>>(
        hout, mapi, bs_p, lfeat, TM, Nn, REP);
}
